// NeuralCausalModel_89653147337501
// MI455X (gfx1250) — matrix operand order repaired and checked by reading, not compiled, not run on hardware
//
#include <hip/hip_runtime.h>
#include <hip/hip_bf16.h>

// ---- types matching the gfx1250 WMMA builtin signatures ----
typedef __bf16 bf16_t;
typedef bf16_t v16bf __attribute__((ext_vector_type(16)));
typedef bf16_t v8bf  __attribute__((ext_vector_type(8)));
typedef bf16_t v4bf  __attribute__((ext_vector_type(4)));
typedef float  v8f   __attribute__((ext_vector_type(8)));

#define NV   256   // num variables
#define ND   512   // hidden dim
#define NB   1024  // batch
#define BM   32    // batch rows per workgroup
#define TPB  256   // 8 wave32

// Build a bf16 B-fragment (32x16, K x N) from 16 contiguous f32 (unaligned-safe, scalar loads)
static __device__ __forceinline__ v16bf cvt_b16_scalar(const float* __restrict__ p) {
    v16bf r;
#pragma unroll
    for (int e = 0; e < 16; ++e) r[e] = (bf16_t)p[e];
    return r;
}

// Same but via four aligned float4 loads (rows stride multiple of 4 floats)
static __device__ __forceinline__ v16bf cvt_b16_vec(const float* __restrict__ p) {
    const float4* q = (const float4*)p;
    float4 f0 = q[0], f1 = q[1], f2 = q[2], f3 = q[3];
    v16bf r;
    r[0]=(bf16_t)f0.x; r[1]=(bf16_t)f0.y; r[2]=(bf16_t)f0.z; r[3]=(bf16_t)f0.w;
    r[4]=(bf16_t)f1.x; r[5]=(bf16_t)f1.y; r[6]=(bf16_t)f1.z; r[7]=(bf16_t)f1.w;
    r[8]=(bf16_t)f2.x; r[9]=(bf16_t)f2.y; r[10]=(bf16_t)f2.z; r[11]=(bf16_t)f2.w;
    r[12]=(bf16_t)f3.x; r[13]=(bf16_t)f3.y; r[14]=(bf16_t)f3.z; r[15]=(bf16_t)f3.w;
    return r;
}

// A-fragment (16x32, M x K, 16-bit) from LDS bf16 row-major [.. x ldk]:
// lane m = lane&15; chunk0 at K = kb + hi*16, chunk1 at K = kb + hi*16 + 8
static __device__ __forceinline__ v16bf load_a_lds(const bf16_t* __restrict__ Ls,
                                                   int row, int ldk, int kb, int hi) {
    const bf16_t* base = Ls + row * ldk + kb + hi * 16;
    v8bf lo = *(const v8bf*)(base);
    v8bf hh = *(const v8bf*)(base + 8);
    return __builtin_shufflevector(lo, hh, 0,1,2,3,4,5,6,7,8,9,10,11,12,13,14,15);
}

__global__ __launch_bounds__(TPB)
void ncm_fused_kernel(const float* __restrict__ x,    // [NB, NV]
                      const float* __restrict__ adj,  // [NV, NV]
                      const float* __restrict__ W1,   // [NV, ND, NV+1]
                      const float* __restrict__ b1,   // [NV, ND]
                      const float* __restrict__ W2,   // [NV, ND, ND]
                      const float* __restrict__ b2,   // [NV, ND]
                      const float* __restrict__ W3,   // [NV, ND]
                      const float* __restrict__ b3,   // [NV]
                      float* __restrict__ out)        // [NB, NV]
{
    __shared__ __align__(16) bf16_t Xs[BM * NV];    // 16 KB: x tile, bf16
    __shared__ __align__(16) bf16_t H1s[BM * ND];   // 32 KB: layer-1 activations, bf16
    __shared__ float wxs[BM];                       // wx column for this variable
    __shared__ float outs[BM];                      // final per-row reduction

    const int t  = threadIdx.x;
    const int i  = blockIdx.y;          // variable index
    const int b0 = blockIdx.x * BM;     // batch tile base

    // ---------- Phase 0: stage x tile as bf16; compute wx[:, i]; zero reduction ----------
    {
        const float4* xp = (const float4*)(x + (size_t)b0 * NV);  // 32 rows contiguous
#pragma unroll
        for (int k = 0; k < (BM * NV / 4) / TPB; ++k) {           // 8 float4 per thread
            int f = t + k * TPB;
            float4 v = xp[f];
            v4bf h; h[0]=(bf16_t)v.x; h[1]=(bf16_t)v.y; h[2]=(bf16_t)v.z; h[3]=(bf16_t)v.w;
            *(v4bf*)&Xs[f * 4] = h;
        }
    }
    if (t < BM) {
        const float4* xr = (const float4*)(x + (size_t)(b0 + t) * NV);
        const float4* ar = (const float4*)(adj + (size_t)i * NV);
        float s = 0.f;
#pragma unroll 4
        for (int j = 0; j < NV / 4; ++j) {
            float4 a = xr[j], b = ar[j];
            s += a.x*b.x + a.y*b.y + a.z*b.z + a.w*b.w;
        }
        wxs[t]  = s;
        outs[t] = 0.f;
    }
    __syncthreads();

    const int lane = t & 31;
    const int wv   = t >> 5;        // wave id 0..7
    const int lr   = lane & 15;
    const int hi   = lane >> 4;
    const int n0   = wv * 64;       // this wave's 64-column slab of D

    const v8f zero = {};
    v8f acc[2][4];                  // 2 M-tiles x 4 N-tiles of 16x16 f32
#pragma unroll
    for (int mt = 0; mt < 2; ++mt)
#pragma unroll
        for (int nt = 0; nt < 4; ++nt) acc[mt][nt] = zero;

    // ---------- Layer 1: h1 = relu([x | wx] @ W1[i]^T + b1[i]) ----------
    // K = 0..255 via WMMA; K = 256 (wx column) as a rank-1 VALU correction.
#pragma unroll 1
    for (int kb = 0; kb < NV; kb += 32) {
        v16bf a[2];
#pragma unroll
        for (int mt = 0; mt < 2; ++mt)
            a[mt] = load_a_lds(Xs, mt * 16 + lr, NV, kb, hi);
#pragma unroll
        for (int nt = 0; nt < 4; ++nt) {
            const int n = n0 + nt * 16 + lr;
            const float* bp = W1 + ((size_t)(i * ND + n)) * (NV + 1) + kb + hi * 16;
            v16bf bfrag = cvt_b16_scalar(bp);   // stride 257 floats -> unaligned, scalar cvt
#pragma unroll
            for (int mt = 0; mt < 2; ++mt)
                acc[mt][nt] = __builtin_amdgcn_wmma_f32_16x16x32_bf16(
                    false, a[mt], false, bfrag, (short)0, acc[mt][nt], false, false);
        }
    }

    // rank-1 wx correction + bias + relu, then park h1 in LDS as bf16
    {
        float wxr[2][8];
#pragma unroll
        for (int mt = 0; mt < 2; ++mt)
#pragma unroll
            for (int v = 0; v < 8; ++v) wxr[mt][v] = wxs[mt * 16 + v + 8 * hi];

#pragma unroll
        for (int nt = 0; nt < 4; ++nt) {
            const int n = n0 + nt * 16 + lr;
            const float w1l  = W1[((size_t)(i * ND + n)) * (NV + 1) + NV]; // K=256 column
            const float bias = b1[(size_t)i * ND + n];
#pragma unroll
            for (int mt = 0; mt < 2; ++mt)
#pragma unroll
                for (int v = 0; v < 8; ++v) {
                    float hv = acc[mt][nt][v] + wxr[mt][v] * w1l + bias;
                    hv = hv > 0.f ? hv : 0.f;
                    H1s[(size_t)(mt * 16 + v + 8 * hi) * ND + n] = (bf16_t)hv;
                }
        }
    }
    __syncthreads();

    // ---------- Layer 2: h2 = relu(h1 @ W2[i]^T + b2[i]), fused with layer 3 ----------
#pragma unroll
    for (int mt = 0; mt < 2; ++mt)
#pragma unroll
        for (int nt = 0; nt < 4; ++nt) acc[mt][nt] = zero;

#pragma unroll 1
    for (int kb = 0; kb < ND; kb += 32) {
        v16bf a[2];
#pragma unroll
        for (int mt = 0; mt < 2; ++mt)
            a[mt] = load_a_lds(H1s, mt * 16 + lr, ND, kb, hi);
#pragma unroll
        for (int nt = 0; nt < 4; ++nt) {
            const int n = n0 + nt * 16 + lr;
            const float* bp = W2 + ((size_t)(i * ND + n)) * ND + kb + hi * 16;
            __builtin_prefetch((const char*)bp + 128, 0, 1);   // next K-chunk of this row
            v16bf bfrag = cvt_b16_vec(bp);                     // stride 512 floats -> float4 aligned
#pragma unroll
            for (int mt = 0; mt < 2; ++mt)
                acc[mt][nt] = __builtin_amdgcn_wmma_f32_16x16x32_bf16(
                    false, a[mt], false, bfrag, (short)0, acc[mt][nt], false, false);
        }
    }

    // bias + relu + dot with W3[i] -> per-row partials; reduce via LDS float atomics
    {
        float tacc[2][8] = {};
#pragma unroll
        for (int nt = 0; nt < 4; ++nt) {
            const int n = n0 + nt * 16 + lr;
            const float bias = b2[(size_t)i * ND + n];
            const float w3v  = W3[(size_t)i * ND + n];
#pragma unroll
            for (int mt = 0; mt < 2; ++mt)
#pragma unroll
                for (int v = 0; v < 8; ++v) {
                    float hv = acc[mt][nt][v] + bias;
                    hv = hv > 0.f ? hv : 0.f;
                    tacc[mt][v] += hv * w3v;
                }
        }
#pragma unroll
        for (int mt = 0; mt < 2; ++mt)
#pragma unroll
            for (int v = 0; v < 8; ++v)
                atomicAdd(&outs[mt * 16 + v + 8 * hi], tacc[mt][v]);
    }
    __syncthreads();

    if (t < BM)
        out[(size_t)(b0 + t) * NV + i] = outs[t] + b3[i];
}

extern "C" void kernel_launch(void* const* d_in, const int* in_sizes, int n_in,
                              void* d_out, int out_size, void* d_ws, size_t ws_size,
                              hipStream_t stream) {
    (void)in_sizes; (void)n_in; (void)d_ws; (void)ws_size; (void)out_size;
    const float* x   = (const float*)d_in[0];
    const float* adj = (const float*)d_in[1];
    const float* W1  = (const float*)d_in[2];
    const float* b1  = (const float*)d_in[3];
    const float* W2  = (const float*)d_in[4];
    const float* b2  = (const float*)d_in[5];
    const float* W3  = (const float*)d_in[6];
    const float* b3  = (const float*)d_in[7];
    float* out = (float*)d_out;

    dim3 grid(NB / BM, NV);   // (32 batch tiles, 256 variables)
    ncm_fused_kernel<<<grid, TPB, 0, stream>>>(x, adj, W1, b1, W2, b2, W3, b3, out);
}
